// DHSEncoderLayer_29128468201803
// MI455X (gfx1250) — matrix operand order repaired and checked by reading, not compiled, not run on hardware
//
#include <hip/hip_runtime.h>
#include <hip/hip_bf16.h>
#include <math.h>

// ---------------- types ----------------
typedef __attribute__((ext_vector_type(16))) _Float16 v16h;
typedef __attribute__((ext_vector_type(8)))  float    v8f;
typedef __attribute__((ext_vector_type(8)))  float    float8;

#define B_   4
#define S_   2048
#define D_   128
#define H_   4
#define V_   64
#define LN_EPS 1e-5f
#define SCALE_ 0.17677669529663687f   /* 1/sqrt(32) */
#define APO_ 0.2f
#define TRP_ 0.15f

// =====================================================================
// Generic WMMA GEMM:  C = alpha*(A @ B^T) [+ bias[n]] [+ bscal[z]] [+ res]
// A: [.., M, K] row-major (f32, converted to f16 in-register)
// B: [.., N, K] row-major (f32 -> f16)        (i.e. computes X @ W^T)
// Per-operand batch offset: (z/div)*outer + (z%div)*inner
// One wave computes a (16*MT) x (16*NT) tile: MT A-fragments are reused
// across NT B-fragments -> MT*NT independent v_wmma_f32_16x16x32_f16 per
// K-step (pipelines in the XDL unit; cuts operand traffic by NT x / MT x).
// =====================================================================
template<int MT, int NT>
__global__ __launch_bounds__(32) void wmma_gemm_kernel(
    const float* __restrict__ A, long aOut, long aIn, int aDiv, int lda,
    const float* __restrict__ Bm, long bOut, long bIn, int bDiv, int ldb,
    float* C, long cOut, long cIn, int cDiv, int ldc,
    const float* __restrict__ bias, const float* __restrict__ bscal,
    const float* res,
    float alpha, int M, int N, int K)
{
    int z = blockIdx.z;
    const float* Ab = A  + (long)(z / aDiv) * aOut + (long)(z % aDiv) * aIn;
    const float* Bb = Bm + (long)(z / bDiv) * bOut + (long)(z % bDiv) * bIn;
    long cbase = (long)(z / cDiv) * cOut + (long)(z % cDiv) * cIn;

    int lane = threadIdx.x;        // one wave
    int row  = lane & 15;
    int hs   = lane >> 4;          // half-wave select

    int m0 = blockIdx.y * (16 * MT);
    int n0 = blockIdx.x * (16 * NT);

    v8f acc[MT][NT] = {};

    for (int k0 = 0; k0 < K; k0 += 32) {
        v16h a[MT] = {};
        v16h b[NT] = {};
        #pragma unroll
        for (int mi = 0; mi < MT; ++mi) {
            int am = m0 + mi * 16 + row;
            if (am < M) {
                // A 16-bit fragment: lanes 0-15 hold K [k0..k0+7] & [k0+16..k0+23],
                // lanes 16-31 hold K [k0+8..k0+15] & [k0+24..k0+31]  (ISA 7.12.2)
                const float* ap = Ab + (long)am * lda + k0 + hs * 8;
                float8 lo = *(const float8*)(ap);
                float8 hi = *(const float8*)(ap + 16);
                #pragma unroll
                for (int i = 0; i < 8; ++i) {
                    a[mi][i] = (_Float16)lo[i]; a[mi][i + 8] = (_Float16)hi[i];
                }
                if (k0 + 32 < K) __builtin_prefetch(ap + 32, 0, 0); // global_prefetch_b8
            }
        }
        #pragma unroll
        for (int ni = 0; ni < NT; ++ni) {
            int bn = n0 + ni * 16 + row;
            if (bn < N) {
                // B fragment (32x16): lanes 0-15 hold K [k0..k0+7] & [k0+16..k0+23] of column n,
                // lanes 16-31 hold K [k0+8..k0+15] & [k0+24..k0+31]
                const float* bp = Bb + (long)bn * ldb + k0 + hs * 8;
                float8 b0v = *(const float8*)(bp);
                float8 b1v = *(const float8*)(bp + 16);
                #pragma unroll
                for (int i = 0; i < 8; ++i) {
                    b[ni][i] = (_Float16)b0v[i]; b[ni][i + 8] = (_Float16)b1v[i];
                }
                if (k0 + 32 < K) __builtin_prefetch(bp + 32, 0, 0);
            }
        }
        // EXEC is all-ones here (divergence above has re-converged).
        // MT*NT independent WMMAs -> back-to-back issue, no RAW hazards.
        #pragma unroll
        for (int mi = 0; mi < MT; ++mi)
            #pragma unroll
            for (int ni = 0; ni < NT; ++ni)
                acc[mi][ni] = __builtin_amdgcn_wmma_f32_16x16x32_f16(
                    false, a[mi], false, b[ni], (short)0, acc[mi][ni], false, false);
    }

    float addc = bscal ? bscal[z] : 0.f;
    #pragma unroll
    for (int ni = 0; ni < NT; ++ni) {
        int cn = n0 + ni * 16 + row;
        if (cn < N) {
            float bb = bias ? bias[cn] : 0.f;
            #pragma unroll
            for (int mi = 0; mi < MT; ++mi) {
                int cmb = m0 + mi * 16 + hs * 8;
                #pragma unroll
                for (int r = 0; r < 8; ++r) {   // C/D: VGPR r -> row cmb+r, col cn
                    int cm = cmb + r;
                    if (cm < M) {
                        long idx = cbase + (long)cm * ldc + cn;
                        float vv = alpha * acc[mi][ni][r] + bb + addc;
                        if (res) vv += res[idx];
                        C[idx] = vv;
                    }
                }
            }
        }
    }
}

// ---------------- LayerNorm over D=128, one block (128 thr) per row ----------------
__global__ void ln_kernel(const float* __restrict__ x, const float* __restrict__ g,
                          const float* __restrict__ b, float* __restrict__ y)
{
    int t = threadIdx.x;
    long base = (long)blockIdx.x * D_;
    __shared__ float red[D_];
    float v = x[base + t];
    red[t] = v; __syncthreads();
    for (int s = 64; s > 0; s >>= 1) { if (t < s) red[t] += red[t + s]; __syncthreads(); }
    float mean = red[0] * (1.0f / D_);
    __syncthreads();
    float d = v - mean;
    red[t] = d * d; __syncthreads();
    for (int s = 64; s > 0; s >>= 1) { if (t < s) red[t] += red[t + s]; __syncthreads(); }
    float var = red[0] * (1.0f / D_);
    y[base + t] = d / sqrtf(var + LN_EPS) * g[t] + b[t];
}

// ---------------- mean over S: obs[b,d] = mean_s in[b,s,d] ----------------
__global__ void mean_s_kernel(const float* __restrict__ xn, float* __restrict__ obs)
{
    int b = blockIdx.x, t = threadIdx.x;   // block 128
    const float* p = xn + (long)b * S_ * D_ + t;
    float s = 0.f;
    for (int i = 0; i < S_; ++i) s += p[(long)i * D_];
    obs[b * D_ + t] = s * (1.0f / S_);
}

// ---------------- infl[b*H+h] = obs[b]·Wobs[h] + bobs[h] ----------------
__global__ void infl_kernel(const float* __restrict__ obs, const float* __restrict__ Wobs,
                            const float* __restrict__ bobs, float* __restrict__ infl)
{
    int t = threadIdx.x;
    if (t < B_ * H_) {
        int b = t >> 2, h = t & 3;
        float s = bobs[h];
        for (int d = 0; d < D_; ++d) s += obs[b * D_ + d] * Wobs[h * D_ + d];
        infl[t] = s;
    }
}

// ---------------- row softmax, in place; one block (256 thr) per row ----------------
__global__ void softmax_kernel(float* __restrict__ p_, int cols)
{
    float* p = p_ + (long)blockIdx.x * cols;
    int t = threadIdx.x;
    __shared__ float red[256];
    float mx = -1e30f;
    for (int c = t; c < cols; c += 256) mx = fmaxf(mx, p[c]);
    red[t] = mx; __syncthreads();
    for (int s = 128; s > 0; s >>= 1) { if (t < s) red[t] = fmaxf(red[t], red[t + s]); __syncthreads(); }
    mx = red[0]; __syncthreads();
    float sum = 0.f;
    for (int c = t; c < cols; c += 256) { float e = expf(p[c] - mx); p[c] = e; sum += e; }
    red[t] = sum; __syncthreads();
    for (int s = 128; s > 0; s >>= 1) { if (t < s) red[t] += red[t + s]; __syncthreads(); }
    float inv = 1.0f / red[0];
    for (int c = t; c < cols; c += 256) p[c] *= inv;
}

// ---------------- vT[(b*H+h)][n][s] = v[b,s,h*32+n] ----------------
__global__ void transpose_v_kernel(const float* __restrict__ v, float* __restrict__ vT, long n)
{
    long i = (long)blockIdx.x * 256 + threadIdx.x;
    if (i >= n) return;
    int s = (int)(i & (S_ - 1));
    long r = i >> 11;
    int nidx = (int)(r & 31);
    int z = (int)(r >> 5);
    int b = z >> 2, h = z & 3;
    vT[i] = v[((long)(b * S_ + s)) * D_ + h * 32 + nidx];
}

__global__ void concat2_kernel(const float* __restrict__ a, const float* __restrict__ b,
                               float* __restrict__ dst, long n)
{
    long i = (long)blockIdx.x * 256 + threadIdx.x;
    if (i >= n) return;
    long row = i >> 8;  int c = (int)(i & 255);
    dst[i] = (c < D_) ? a[row * D_ + c] : b[row * D_ + (c - D_)];
}

__global__ void concat3_kernel(const float* __restrict__ tv, const float* __restrict__ cv,
                               const float* __restrict__ sv, float* __restrict__ dst, long n)
{
    long i = (long)blockIdx.x * 256 + threadIdx.x;
    if (i >= n) return;
    long row = i / 192;  int c = (int)(i % 192);
    int b = (int)(row / S_);
    float v;
    if (c < V_)            v = tv[row * V_ + c];
    else if (c < 2 * V_)   v = cv[b * V_ + (c - V_)];
    else                   v = sv[row * V_ + (c - 2 * V_)];
    dst[i] = v;
}

__global__ void gate_update_kernel(float* cur, const float* __restrict__ out,
                                   const float* __restrict__ gpre, long n)
{
    long i = (long)blockIdx.x * 256 + threadIdx.x;
    if (i >= n) return;
    float g = 1.0f / (1.0f + expf(-gpre[i]));
    cur[i] = cur[i] * (1.0f - g) + out[i] * g;
}

__global__ void add2_kernel(float* __restrict__ dst, const float* __restrict__ a,
                            const float* __restrict__ b, long n)
{
    long i = (long)blockIdx.x * 256 + threadIdx.x;
    if (i >= n) return;
    dst[i] = a[i] + b[i];
}

__global__ void cvnorm_kernel(const float* __restrict__ cv, float* __restrict__ cn)
{
    int b = blockIdx.x, t = threadIdx.x;   // block 64
    __shared__ float red[V_];
    float v = cv[b * V_ + t];
    red[t] = v * v; __syncthreads();
    for (int s = 32; s > 0; s >>= 1) { if (t < s) red[t] += red[t + s]; __syncthreads(); }
    if (t == 0) cn[b] = sqrtf(red[0]);
}

__global__ void trp_resolve_kernel(const float* __restrict__ tv, const float* __restrict__ cv,
                                   const float* __restrict__ sv, const float* __restrict__ rg,
                                   const float* __restrict__ cn, const float* __restrict__ fv,
                                   float* __restrict__ resolved)
{
    long row = blockIdx.x;                 // B*S rows, block 64
    int b = (int)(row / S_);
    int t = threadIdx.x;
    __shared__ float sd[V_], ss[V_];
    long base = row * V_;
    float tvv = tv[base + t];
    float svv = sv[base + t];
    float cvv = cv[b * V_ + t];
    sd[t] = cvv * svv; ss[t] = svv * svv;
    __syncthreads();
    for (int s = 32; s > 0; s >>= 1) {
        if (t < s) { sd[t] += sd[t + s]; ss[t] += ss[t + s]; }
        __syncthreads();
    }
    float align = sd[0] / (fmaxf(cn[b], 1e-8f) * fmaxf(sqrtf(ss[0]), 1e-8f));
    float r = 1.0f / (1.0f + expf(-rg[base + t]));
    resolved[base + t] = tvv + TRP_ * (-align * (fv[t] - tvv) * r);
}

__global__ void gelu_kernel(float* p, long n)
{
    long i = (long)blockIdx.x * 256 + threadIdx.x;
    if (i >= n) return;
    float v = p[i];
    p[i] = 0.5f * v * (1.0f + erff(v * 0.7071067811865476f));
}

__global__ void ffn_update_kernel(float* x, const float* __restrict__ ff,
                                  const float* __restrict__ sg, long n)
{
    long i = (long)blockIdx.x * 256 + threadIdx.x;
    if (i >= n) return;
    x[i] += (1.0f / (1.0f + expf(-sg[i]))) * ff[i];
}

__global__ void apo_update_kernel(float* __restrict__ o, const float* __restrict__ x,
                                  const float* __restrict__ tp, long n)
{
    long i = (long)blockIdx.x * 256 + threadIdx.x;
    if (i >= n) return;
    o[i] = x[i] - APO_ * tanhf(tp[i]);
}

// =====================================================================
extern "C" void kernel_launch(void* const* d_in, const int* in_sizes, int n_in,
                              void* d_out, int out_size, void* d_ws, size_t ws_size,
                              hipStream_t stream)
{
    (void)in_sizes; (void)n_in; (void)out_size; (void)ws_size;
    const float* x_in  = (const float*)d_in[0];
    const float* space = (const float*)d_in[1];
    const float* g_attn = (const float*)d_in[2];
    const float* b_attn = (const float*)d_in[3];
    const float* Wq = (const float*)d_in[4];
    const float* Wk = (const float*)d_in[5];
    const float* Wv = (const float*)d_in[6];
    const float* Wo = (const float*)d_in[7];
    const float* bo = (const float*)d_in[8];
    const float* Wg = (const float*)d_in[9];
    const float* bg = (const float*)d_in[10];
    const float* Wobs = (const float*)d_in[11];
    const float* bobs = (const float*)d_in[12];
    const float* g_trp = (const float*)d_in[13];
    const float* b_trp = (const float*)d_in[14];
    const float* Wt = (const float*)d_in[15];
    const float* bt = (const float*)d_in[16];
    const float* Wc = (const float*)d_in[17];
    const float* bc = (const float*)d_in[18];
    const float* Ws = (const float*)d_in[19];
    const float* bs = (const float*)d_in[20];
    const float* Wr = (const float*)d_in[21];
    const float* br = (const float*)d_in[22];
    const float* fv = (const float*)d_in[23];
    const float* Wop = (const float*)d_in[24];
    const float* bop = (const float*)d_in[25];
    const float* g_ffn = (const float*)d_in[26];
    const float* b_ffn = (const float*)d_in[27];
    const float* W1 = (const float*)d_in[28];
    const float* b1 = (const float*)d_in[29];
    const float* W2 = (const float*)d_in[30];
    const float* b2 = (const float*)d_in[31];
    const float* Wsg = (const float*)d_in[32];
    const float* bsg = (const float*)d_in[33];
    const float* g_apo = (const float*)d_in[34];
    const float* b_apo = (const float*)d_in[35];
    const float* Wn = (const float*)d_in[36];
    const float* bn = (const float*)d_in[37];

    const long BS  = (long)B_ * S_;          // 8192
    const long BSD = BS * D_;                // 1048576
    const long SS  = (long)S_ * S_;          // 4194304

    float* out_x = (float*)d_out;            // [B,S,D]
    float* aw    = out_x + BSD;              // [B,H,S,S] — score scratch + final output

    // ---- workspace carve ----
    float* wsf = (float*)d_ws;
    size_t off = 0;
    auto alloc = [&](size_t nelem) {
        float* p = wsf + off;
        off += (nelem + 63) & ~(size_t)63;
        return p;
    };
    float* xn    = alloc(BSD);   // ln outputs (reused for xn2/xn3/xn4)
    float* obs   = alloc(512);   // also obs2
    float* infl  = alloc(64);
    float* cur   = alloc(BSD);
    float* qb    = alloc(BSD);   // q, then ao, then tv, then ff
    float* kb    = alloc(BSD);   // k, then sv, then sgpre
    float* vb    = alloc(BSD);   // v, then rgpre, then tanh-pre
    float* vT    = alloc(BSD);
    float* outb  = alloc(BSD);   // attn out, then resolved
    float* cat2  = alloc(BS * 2 * D_); // also cat3
    float* gpre  = alloc(BSD);
    float* xw    = alloc(BSD);   // running residual stream
    float* cvb   = alloc(256);
    float* cnb   = alloc(64);
    float* ffpre = alloc(BS * 4 * D_);

    auto gemm = [&](int MTs, int NTs,
                    const float* A, long aOut, long aIn, int aDiv, int lda,
                    const float* Bm, long bOut, long bIn, int bDiv, int ldb,
                    float* C, long cOut, long cIn, int cDiv, int ldc,
                    const float* bias, const float* bscal, const float* res,
                    float alpha, int M, int N, int K, int batches) {
        dim3 grid((N + 16 * NTs - 1) / (16 * NTs), (M + 16 * MTs - 1) / (16 * MTs), batches);
        if (MTs == 2 && NTs == 4)
            wmma_gemm_kernel<2,4><<<grid, dim3(32), 0, stream>>>(
                A,aOut,aIn,aDiv,lda, Bm,bOut,bIn,bDiv,ldb, C,cOut,cIn,cDiv,ldc,
                bias,bscal,res, alpha,M,N,K);
        else if (MTs == 2 && NTs == 2)
            wmma_gemm_kernel<2,2><<<grid, dim3(32), 0, stream>>>(
                A,aOut,aIn,aDiv,lda, Bm,bOut,bIn,bDiv,ldb, C,cOut,cIn,cDiv,ldc,
                bias,bscal,res, alpha,M,N,K);
        else
            wmma_gemm_kernel<1,4><<<grid, dim3(32), 0, stream>>>(
                A,aOut,aIn,aDiv,lda, Bm,bOut,bIn,bDiv,ldb, C,cOut,cIn,cDiv,ldc,
                bias,bscal,res, alpha,M,N,K);
    };
    auto ew = [&](long n) { return dim3((unsigned)((n + 255) / 256)); };

    // ================= recursive attention =================
    ln_kernel<<<(unsigned)BS, D_, 0, stream>>>(x_in, g_attn, b_attn, xn);
    mean_s_kernel<<<B_, D_, 0, stream>>>(xn, obs);
    infl_kernel<<<1, 32, 0, stream>>>(obs, Wobs, bobs, infl);
    hipMemcpyAsync(cur, xn, BSD * sizeof(float), hipMemcpyDeviceToDevice, stream);

    for (int it = 0; it < 3; ++it) {
        gemm(2,4, cur,0,0,1,D_, Wq,0,0,1,D_, qb,0,0,1,D_, nullptr,nullptr,nullptr, 1.f, (int)BS, D_, D_, 1);
        gemm(2,4, cur,0,0,1,D_, Wk,0,0,1,D_, kb,0,0,1,D_, nullptr,nullptr,nullptr, 1.f, (int)BS, D_, D_, 1);
        gemm(2,4, cur,0,0,1,D_, Wv,0,0,1,D_, vb,0,0,1,D_, nullptr,nullptr,nullptr, 1.f, (int)BS, D_, D_, 1);
        transpose_v_kernel<<<ew(BSD), 256, 0, stream>>>(vb, vT, BSD);

        // scores[b,h] = SCALE * q[b,h] @ k[b,h]^T + infl[b,h]   (K = 32)
        gemm(2,4, qb, (long)S_*D_, 32, H_, D_,
             kb, (long)S_*D_, 32, H_, D_,
             aw, (long)H_*SS, SS, H_, S_,
             nullptr, infl, nullptr, SCALE_, S_, S_, 32, B_*H_);
        softmax_kernel<<<(unsigned)(B_*H_*S_), 256, 0, stream>>>(aw, S_);

        // ao[b,h] = aw[b,h] @ v[b,h]   via vT,  ao written head-interleaved into qb
        gemm(2,2, aw, (long)H_*SS, SS, H_, S_,
             vT, 32L*S_, 0, 1, S_,
             qb, (long)S_*D_, 32, H_, D_,
             nullptr, nullptr, nullptr, 1.f, S_, 32, S_, B_*H_);

        gemm(2,4, qb,0,0,1,D_, Wo,0,0,1,D_, outb,0,0,1,D_, bo,nullptr,nullptr, 1.f, (int)BS, D_, D_, 1);
        concat2_kernel<<<ew(BS*2*D_), 256, 0, stream>>>(cur, outb, cat2, BS*2*D_);
        gemm(2,4, cat2,0,0,1,2*D_, Wg,0,0,1,2*D_, gpre,0,0,1,D_, bg,nullptr,nullptr, 1.f, (int)BS, D_, 2*D_, 1);
        gate_update_kernel<<<ew(BSD), 256, 0, stream>>>(cur, outb, gpre, BSD);
    }
    add2_kernel<<<ew(BSD), 256, 0, stream>>>(xw, x_in, cur, BSD);

    // ================= trauma resolution path =================
    ln_kernel<<<(unsigned)BS, D_, 0, stream>>>(xw, g_trp, b_trp, xn);
    mean_s_kernel<<<B_, D_, 0, stream>>>(xn, obs);
    float* tv = qb; float* sv = kb; float* rgpre = vb; float* cat3 = cat2; float* resolved = outb;
    gemm(2,4, xn,0,0,1,D_,  Wt,0,0,1,D_, tv,0,0,1,V_,  bt,nullptr,nullptr, 1.f, (int)BS, V_, D_, 1);
    gemm(1,4, obs,0,0,1,D_, Wc,0,0,1,D_, cvb,0,0,1,V_, bc,nullptr,nullptr, 1.f, B_, V_, D_, 1);
    cvnorm_kernel<<<B_, V_, 0, stream>>>(cvb, cnb);
    gemm(2,4, space,0,0,1,D_, Ws,0,0,1,D_, sv,0,0,1,V_, bs,nullptr,nullptr, 1.f, (int)BS, V_, D_, 1);
    concat3_kernel<<<ew(BS*3*V_), 256, 0, stream>>>(tv, cvb, sv, cat3, BS*3*V_);
    gemm(2,4, cat3,0,0,1,3*V_, Wr,0,0,1,3*V_, rgpre,0,0,1,V_, br,nullptr,nullptr, 1.f, (int)BS, V_, 3*V_, 1);
    trp_resolve_kernel<<<(unsigned)BS, V_, 0, stream>>>(tv, cvb, sv, rgpre, cnb, fv, resolved);
    gemm(2,4, resolved,0,0,1,V_, Wop,0,0,1,V_, xw,0,0,1,D_, bop,nullptr,xw, 1.f, (int)BS, D_, V_, 1);

    // ================= morphogenesis FFN =================
    ln_kernel<<<(unsigned)BS, D_, 0, stream>>>(xw, g_ffn, b_ffn, xn);
    gemm(2,4, xn,0,0,1,D_, W1,0,0,1,D_, ffpre,0,0,1,4*D_, b1,nullptr,nullptr, 1.f, (int)BS, 4*D_, D_, 1);
    gelu_kernel<<<ew(BS*4*D_), 256, 0, stream>>>(ffpre, BS*4*D_);
    float* ff = qb; float* sgpre = kb;
    gemm(2,4, ffpre,0,0,1,4*D_, W2,0,0,1,4*D_, ff,0,0,1,D_, b2,nullptr,nullptr, 1.f, (int)BS, D_, 4*D_, 1);
    gemm(2,4, xn,0,0,1,D_, Wsg,0,0,1,D_, sgpre,0,0,1,D_, bsg,nullptr,nullptr, 1.f, (int)BS, D_, D_, 1);
    ffn_update_kernel<<<ew(BSD), 256, 0, stream>>>(xw, ff, sgpre, BSD);

    // ================= apophasis =================
    ln_kernel<<<(unsigned)BS, D_, 0, stream>>>(xw, g_apo, b_apo, xn);
    float* tpre = vb;
    gemm(2,4, xn,0,0,1,D_, Wn,0,0,1,D_, tpre,0,0,1,D_, bn,nullptr,nullptr, 1.f, (int)BS, D_, D_, 1);
    apo_update_kernel<<<ew(BSD), 256, 0, stream>>>(out_x, xw, tpre, BSD);
}
